// DynamicconvLayer_52407190945863
// MI455X (gfx1250) — matrix operand order repaired and checked by reading, not compiled, not run on hardware
//
#include <hip/hip_runtime.h>
#include <hip/hip_bf16.h>

// Problem constants (reference: T,B,C = 2048,8,1024; H,K,P = 16,31,15)
#define Tt 2048
#define Bt 8
#define Ct 1024
#define Ht 16
#define KT 31
#define Pt 15
#define Ot (Ht * KT)      // 496 logits per row (= 31 tiles of 16)
#define Mt (Tt * Bt)      // 16384 GEMM rows
#define Rt (Ct / Ht)      // 64 channels per head

typedef __attribute__((ext_vector_type(16))) __bf16 v16bf;
typedef __attribute__((ext_vector_type(8)))  float  v8f;

union BF16Frag { v16bf v; uint4 q[2]; };

__device__ __forceinline__ unsigned bf16pk(float a, float b) {
    unsigned ua = __builtin_bit_cast(unsigned, a);
    unsigned ub = __builtin_bit_cast(unsigned, b);
    // round-to-nearest-even bf16
    ua = (ua + 0x7FFFu + ((ua >> 16) & 1u)) >> 16;
    ub = (ub + 0x7FFFu + ((ub >> 16) & 1u)) >> 16;
    return (ua & 0xFFFFu) | (ub << 16);
}

// ---------------------------------------------------------------------------
// Kernel 1: W (O=496, C=1024) f32 -> bf16 (same row-major layout)
// ---------------------------------------------------------------------------
__global__ __launch_bounds__(256)
void wconv_bf16(const float* __restrict__ W, __bf16* __restrict__ Wb) {
    int i = (blockIdx.x * 256 + threadIdx.x) * 4;     // Ot*Ct = 507904
    if (i < Ot * Ct) {
        float4 f = *(const float4*)(W + i);
        uint2 u;
        u.x = bf16pk(f.x, f.y);
        u.y = bf16pk(f.z, f.w);
        *(uint2*)(Wb + i) = u;
    }
}

// ---------------------------------------------------------------------------
// Kernel 2: logits = x(16384,1024) @ Wb^T(1024,496), v_wmma_f32_16x16x32_bf16.
// 512 threads = 16 waves (4/SIMD). Wave (msub, njgroup): 16-row M-subtile x
// 8 N-tiles, persistent v8f acc[8]. Global->reg->LDS pipeline: issue loads for
// slab ks+1, run 8 WMMAs on slab ks from LDS, then convert+commit ks+1, so the
// s_wait_loadcnt lands AFTER the matrix work and global latency is overlapped.
// ---------------------------------------------------------------------------
#define BM 64
#define SLDA 40   // padded slab row stride in bf16 (20 dwords -> conflict-free b128)

__global__ __launch_bounds__(512)
void gemm_logits(const float* __restrict__ x, const __bf16* __restrict__ Wb,
                 float* __restrict__ logits) {
    __shared__ __align__(16) __bf16 aS[2][BM * SLDA];   // 2 x  5,120 B
    __shared__ __align__(16) __bf16 bS[2][Ot * SLDA];   // 2 x 39,680 B

    const int tid  = threadIdx.x;
    const int row0 = blockIdx.x * BM;

    const int wave = tid >> 5;            // 0..15
    const int lane = tid & 31;
    const int lm   = lane & 15;
    const int hi   = lane >> 4;           // half-wave select
    const int msub = wave & 3;            // M-subtile 0..3
    const int grp  = wave >> 2;           // N-group 0..3
    const int njlo = (grp < 3) ? grp * 8 : 23;  // {0,8,16,23}; nj=23 duplicated

    v8f acc[8];
    #pragma unroll
    for (int i = 0; i < 8; ++i) acc[i] = (v8f){0.f,0.f,0.f,0.f,0.f,0.f,0.f,0.f};

    // staging registers: A = 1 float4/thread, B = 8 uint2/thread
    float4 av;
    uint2  bv[8];
    const int ar = tid >> 3;              // A slab row (8 float4 per row)
    const int ac = (tid & 7) << 2;        // A slab col

    auto issue = [&](int kk) {            // issue global loads only
        av = *(const float4*)(x + (size_t)(row0 + ar) * Ct + kk + ac);
        #pragma unroll
        for (int j = 0; j < 8; ++j) {
            int idx = tid + j * 512;      // guard is wave-uniform (cut at tid 384)
            if (idx < Ot * 8) {
                int n = idx >> 3, c = (idx & 7) << 2;
                bv[j] = *(const uint2*)(Wb + (size_t)n * Ct + kk + c);
            }
        }
    };
    auto commit = [&](int buf) {          // convert + ds_store (consumes loads)
        uint2 u;
        u.x = bf16pk(av.x, av.y);
        u.y = bf16pk(av.z, av.w);
        *(uint2*)(&aS[buf][ar * SLDA + ac]) = u;
        #pragma unroll
        for (int j = 0; j < 8; ++j) {
            int idx = tid + j * 512;
            if (idx < Ot * 8) {
                int n = idx >> 3, c = (idx & 7) << 2;
                *(uint2*)(&bS[buf][n * SLDA + c]) = bv[j];
            }
        }
    };

    issue(0);
    commit(0);
    __syncthreads();

    for (int ks = 0; ks < Ct / 32; ++ks) {
        const int cur = ks & 1;
        if (ks + 1 < Ct / 32) issue((ks + 1) * 32);   // loads fly during WMMAs

        // A fragment: row = msub*16 + lm, K = 8*hi + {0..7, 16..23}
        BF16Frag a;
        const __bf16* ap = &aS[cur][(msub * 16 + lm) * SLDA + hi * 8];
        a.q[0] = *(const uint4*)(ap);
        a.q[1] = *(const uint4*)(ap + 16);

        // B fragments: col n = (njlo+nj)*16 + lm, K = 8*hi + {0..7, 16..23}
        const __bf16* bp = &bS[cur][(njlo * 16 + lm) * SLDA + hi * 8];
        #pragma unroll
        for (int nj = 0; nj < 8; ++nj) {
            BF16Frag b;
            b.q[0] = *(const uint4*)(bp + nj * 16 * SLDA);
            b.q[1] = *(const uint4*)(bp + nj * 16 * SLDA + 16);
            acc[nj] = __builtin_amdgcn_wmma_f32_16x16x32_bf16(
                false, a.v, false, b.v, (short)0, acc[nj], false, false);
        }

        if (ks + 1 < Ct / 32) commit(cur ^ 1);        // wait_loadcnt lands here
        __syncthreads();
    }

    // D layout: lane lm = column, VGPR i -> M = i + 8*hi
    #pragma unroll
    for (int nj = 0; nj < 8; ++nj) {
        float* outp = logits + (size_t)(row0 + msub * 16 + 8 * hi) * Ot
                             + (njlo + nj) * 16 + lm;
        #pragma unroll
        for (int i = 0; i < 8; ++i)
            outp[(size_t)i * Ot] = acc[nj][i];
    }
}

// ---------------------------------------------------------------------------
// Kernel 3: softmax over K=31 + depthwise dynamic conv (fp32, memory-bound)
// Block = (t-tile of 128) x b x h, 256 threads.
// ---------------------------------------------------------------------------
#define TT 128

__global__ __launch_bounds__(256)
void dynconv(const float* __restrict__ x, const float* __restrict__ logits,
             float* __restrict__ out) {
    __shared__ float xs[(TT + KT - 1) * Rt];   // 158*64*4 = 40448 B halo slab
    __shared__ float wls[TT * 32];             // softmaxed weights, stride 32

    const int tid = threadIdx.x;
    const int t0  = blockIdx.x * TT;
    const int b   = blockIdx.y;
    const int h   = blockIdx.z;

    // load x halo slab: rows t0-P .. t0+TT+K-2-P (zero-padded at edges)
    {
        const int c = tid & (Rt - 1);
        for (int j = tid >> 6; j < TT + KT - 1; j += 4) {
            int t = t0 - Pt + j;
            float v = 0.f;
            if (t >= 0 && t < Tt)
                v = x[(size_t)t * (Bt * Ct) + b * Ct + h * Rt + c];
            xs[j * Rt + c] = v;
        }
    }

    // softmax: threads 0..127 each own one t position
    if (tid < TT) {
        const float* lp = logits + ((size_t)(t0 + tid) * Bt + b) * Ot + h * KT;
        float v[KT];
        float m = -3.4e38f;
        #pragma unroll
        for (int k = 0; k < KT; ++k) { v[k] = lp[k]; m = fmaxf(m, v[k]); }
        float s = 0.f;
        #pragma unroll
        for (int k = 0; k < KT; ++k) { v[k] = __expf(v[k] - m); s += v[k]; }
        float inv = 1.0f / s;
        #pragma unroll
        for (int k = 0; k < KT; ++k) wls[tid * 32 + k] = v[k] * inv;
    }
    __syncthreads();

    // conv: wave-uniform t (LDS broadcast of weights), consecutive r lanes
    const int r  = tid & (Rt - 1);
    const int tq = tid >> 6;
    for (int p = 0; p < TT / 4; ++p) {
        const int tl = tq + 4 * p;
        float s = 0.f;
        #pragma unroll
        for (int k = 0; k < KT; ++k)
            s = fmaf(wls[tl * 32 + k], xs[(tl + k) * Rt + r], s);
        out[(size_t)(t0 + tl) * (Bt * Ct) + b * Ct + h * Rt + r] = s;
    }
}

// ---------------------------------------------------------------------------
extern "C" void kernel_launch(void* const* d_in, const int* in_sizes, int n_in,
                              void* d_out, int out_size, void* d_ws, size_t ws_size,
                              hipStream_t stream) {
    const float* x = (const float*)d_in[0];   // (T,B,C) f32
    const float* W = (const float*)d_in[1];   // (H*K,C) f32
    float* out = (float*)d_out;               // (T,B,C) f32

    // workspace layout: [0, 1MB) Wb bf16 ; [2MB, 2MB+31MB) logits f32
    __bf16* Wb     = (__bf16*)d_ws;
    float*  logits = (float*)((char*)d_ws + (2u << 20));

    wconv_bf16<<<(Ot * Ct / 4 + 255) / 256, 256, 0, stream>>>(W, Wb);
    gemm_logits<<<Mt / BM, 512, 0, stream>>>(x, Wb, logits);
    dim3 g3(Tt / TT, Bt, Ht);
    dynconv<<<g3, 256, 0, stream>>>(x, logits, out);
}
